// CRF_84877143703937
// MI455X (gfx1250) — compile-verified, matrix operand order checked by reading, not run on hardware
//
#include <hip/hip_runtime.h>
#include <hip/hip_bf16.h>

typedef _Float16 v16h __attribute__((ext_vector_type(16)));
typedef _Float16 half4_t __attribute__((ext_vector_type(4)));
typedef float    v8f  __attribute__((ext_vector_type(8)));
typedef float    v4f  __attribute__((ext_vector_type(4)));
typedef int      crf_v4i __attribute__((vector_size(16)));

constexpr int Sdim = 1024;
constexpr int Bdim = 512;
constexpr int NTg  = 64;     // num tags
constexpr int TROW = 66;     // transitions row stride (NT+2)

#define CRF_LOG2E 1.4426950408889634f
#define CRF_LN2   0.6931471805599453f

#if __has_builtin(__builtin_amdgcn_global_load_async_to_lds_b128)
#define CRF_ASYNC_EM 1
#endif

// ---------------------------------------------------------------------------
// Kernel 1: numerator per batch  (em_score + trans_score + end_trans)
// ---------------------------------------------------------------------------
__global__ __launch_bounds__(256)
void CRF_numer_kernel(const float* __restrict__ em,
                      const int* __restrict__ tags,
                      const unsigned char* __restrict__ mask,
                      const float* __restrict__ trans,
                      float* __restrict__ num) {
    const int b   = blockIdx.x;
    const int tid = threadIdx.x;
    __shared__ float redf[256];
    __shared__ int   redi[256];

    const int* tg            = tags + (size_t)b * Sdim;
    const unsigned char* mk  = mask + (size_t)b * Sdim;
    const float* eb          = em   + (size_t)b * Sdim * NTg;

    float acc = 0.0f;
    int   cnt = 0;
    for (int t = tid; t < Sdim; t += 256) {
        int m   = mk[t];
        int tgt = tg[t];
        int pv  = (t == 0) ? NTg : tg[t - 1];   // START = NT
        if (m) {
            acc += eb[(size_t)t * NTg + tgt] + trans[pv * TROW + tgt];
            cnt += 1;
        }
    }
    redf[tid] = acc;
    redi[tid] = cnt;
    __syncthreads();
    for (int s2 = 128; s2 > 0; s2 >>= 1) {
        if (tid < s2) { redf[tid] += redf[tid + s2]; redi[tid] += redi[tid + s2]; }
        __syncthreads();
    }
    if (tid == 0) {
        int len = redi[0];
        int et  = tg[len - 1];
        num[b]  = redf[0] + trans[et * TROW + (NTg + 1)];   // END = NT+1
    }
}

// ---------------------------------------------------------------------------
// Kernel 2: forward algorithm (denominator) via WMMA.
//   grid = 32 workgroups (16 batches each), block = 128 threads = 4 waves.
//   Wave w owns N-tile w of alpha (16 cols), kept in C/D register layout.
//   E = exp(T) kept in B-layout registers for the whole scan.
//   Per step:  new_alpha = em + mx + ln2 * log2( exp2((alpha-mx)*log2e) @ E )
//   Emissions for step t+1 streamed via GLOBAL_LOAD_ASYNC_TO_LDS_B128
//   (double-buffered, ASYNCcnt-synchronized) while step t computes.
//   Pbuf/Red double-buffered on step parity -> only 2 barriers per step.
// ---------------------------------------------------------------------------
__global__ __launch_bounds__(128)
void CRF_forward_kernel(const float* __restrict__ em,
                        const float* __restrict__ trans,
                        const unsigned char* __restrict__ mask,
                        float* __restrict__ den) {
    __shared__ __align__(16) _Float16 Pbuf[2][16][68];  // padded: stride 136B
    __shared__ __align__(16) float    Red[2][16][4];    // per-batch per-wave reduce slots
    __shared__ int lenS[16];
#ifdef CRF_ASYNC_EM
    __shared__ __align__(16) float    Ebuf[2][16][68];  // padded rows (272B)
#endif

    const int tid   = threadIdx.x;
    const int w     = tid >> 5;          // wave id == N-tile id (0..3)
    const int lane  = tid & 31;
    const int half  = lane >> 4;         // 0: M 0-7 / K-lo, 1: M 8-15 / K-hi
    const int lcol  = lane & 15;
    const int bbase = blockIdx.x * 16;
    const int n     = w * 16 + lcol;     // this lane's alpha column

#ifdef CRF_ASYNC_EM
    // Issue the 4KB emission tile for step t into Ebuf[buf]: 256 x 16B chunks,
    // 2 per thread, via the async-to-LDS pipe (tracked on ASYNCcnt).
    auto issue_em = [&](int t, int buf) {
#pragma unroll
        for (int i = 0; i < 2; ++i) {
            int slot = tid + 128 * i;            // 0..255
            int row  = slot >> 4;                // batch row 0..15
            int c4   = slot & 15;                // 16B chunk in row
            const float* g = &em[(size_t)(bbase + row) * Sdim * NTg +
                                 (size_t)t * NTg + c4 * 4];
            float* l = &Ebuf[buf][row][c4 * 4];
            __builtin_amdgcn_global_load_async_to_lds_b128(
                (__attribute__((address_space(1))) crf_v4i*)(uintptr_t)g,
                (__attribute__((address_space(3))) crf_v4i*)(unsigned)(uintptr_t)l,
                0, 0);
        }
    };
#endif

    // ---- sequence lengths (mask is monotone per row) ----
    if (tid < 16) lenS[tid] = 0;
    __syncthreads();
    {
        const int bb  = tid >> 3;
        const int seg = tid & 7;
        const unsigned char* mp = mask + (size_t)(bbase + bb) * Sdim + seg * 128;
        int c = 0;
        for (int i = 0; i < 128; ++i) c += mp[i];
        atomicAdd(&lenS[bb], c);
    }
    __syncthreads();
    int lenr[8];
#pragma unroll
    for (int vr = 0; vr < 8; ++vr) lenr[vr] = lenS[vr + 8 * half];

    // ---- E = exp(T[:64,:64]) resident in B-layout registers (K=64 -> 2 tiles) ----
    v16h Bt0, Bt1;
#pragma unroll
    for (int h = 0; h < 16; ++h) {
        int Kl = (h < 8) ? (half * 8 + h) : (16 + half * 8 + (h - 8));
        Bt0[h] = (_Float16)__expf(trans[(Kl)      * TROW + n]);
        Bt1[h] = (_Float16)__expf(trans[(32 + Kl) * TROW + n]);
    }

    // ---- alpha init: trans[START, n] + emissions[b, 0, n]  (C/D layout) ----
    v8f D;
    const float startTn = trans[NTg * TROW + n];
#pragma unroll
    for (int vr = 0; vr < 8; ++vr) {
        int m = vr + 8 * half;
        D[vr] = startTn + em[(size_t)(bbase + m) * Sdim * NTg + n];
    }

#ifdef CRF_ASYNC_EM
    issue_em(1, 0);                       // prime the pipeline for t=1
#endif

    const int kh = half * 8;

    for (int t = 1; t < Sdim; ++t) {
        const int par = t & 1;
#ifdef CRF_ASYNC_EM
        const int cur = (t - 1) & 1;
        if (t + 1 < Sdim) {
            issue_em(t + 1, par);         // overlap t+1 fetch with step t
            asm volatile("s_wait_asynccnt 2" ::: "memory");  // retire step-t transfers
        } else {
            asm volatile("s_wait_asynccnt 0" ::: "memory");
        }
#else
        float e[8];
#pragma unroll
        for (int vr = 0; vr < 8; ++vr) {
            int m = vr + 8 * half;
            const float* p = &em[(size_t)(bbase + m) * Sdim * NTg + (size_t)t * NTg + n];
            e[vr] = *p;
            if (t + 1 < Sdim) __builtin_prefetch(p + NTg, 0, 0);
        }
#endif

        // per-batch partial max over this wave's 16 columns (stays inside half)
        float pm[8];
#pragma unroll
        for (int vr = 0; vr < 8; ++vr) {
            float v = D[vr];
            v = fmaxf(v, __shfl_xor(v, 1));
            v = fmaxf(v, __shfl_xor(v, 2));
            v = fmaxf(v, __shfl_xor(v, 4));
            v = fmaxf(v, __shfl_xor(v, 8));
            pm[vr] = v;
        }
        if (lcol == 0) {
#pragma unroll
            for (int vr = 0; vr < 8; ++vr) Red[par][vr + 8 * half][w] = pm[vr];
        }
        __syncthreads();   // barrier 1: publishes Red + all async LDS writes for t
        float mx[8];
#pragma unroll
        for (int vr = 0; vr < 8; ++vr) {
            v4f q = *(const v4f*)&Red[par][vr + 8 * half][0];
            mx[vr] = fmaxf(fmaxf(q[0], q[1]), fmaxf(q[2], q[3]));
        }

#ifdef CRF_ASYNC_EM
        float e[8];
#pragma unroll
        for (int vr = 0; vr < 8; ++vr) e[vr] = Ebuf[cur][vr + 8 * half][n];
#endif

        // P = exp2((alpha - mx) * log2e) staged to LDS row-major [m][k] (f16)
#pragma unroll
        for (int vr = 0; vr < 8; ++vr)
            Pbuf[par][vr + 8 * half][n] =
                (_Float16)__builtin_amdgcn_exp2f((D[vr] - mx[vr]) * CRF_LOG2E);
        __syncthreads();   // barrier 2: P tile complete

        // gather A tiles (16x32 each) from LDS in WMMA A-layout
        const _Float16* prow = &Pbuf[par][lcol][0];
        v16h A0{}, A1{};
        {
            auto LD8 = [&](int kbase, v16h& A, int hb) {
                const half4_t* q = (const half4_t*)(prow + kbase);
                half4_t c0 = q[0], c1 = q[1];
                A[hb + 0] = c0[0]; A[hb + 1] = c0[1]; A[hb + 2] = c0[2]; A[hb + 3] = c0[3];
                A[hb + 4] = c1[0]; A[hb + 5] = c1[1]; A[hb + 6] = c1[2]; A[hb + 7] = c1[3];
            };
            LD8(kh,      A0, 0); LD8(16 + kh, A0, 8);
            LD8(32 + kh, A1, 0); LD8(48 + kh, A1, 8);
        }

        // acc = P @ E for this wave's 16x16 output tile (f32 accumulate)
        v8f acc = {};
        acc = __builtin_amdgcn_wmma_f32_16x16x32_f16(false, A0, false, Bt0,
                                                     (short)0, acc, false, false);
        acc = __builtin_amdgcn_wmma_f32_16x16x32_f16(false, A1, false, Bt1,
                                                     (short)0, acc, false, false);

        // alpha update with mask (t < len); acc >= min(E) > 0 so raw v_log is safe
#pragma unroll
        for (int vr = 0; vr < 8; ++vr) {
            float na = e[vr] + mx[vr] + CRF_LN2 * __builtin_amdgcn_logf(acc[vr]);
            D[vr] = (t < lenr[vr]) ? na : D[vr];
        }
        // no end-of-loop barrier: Pbuf/Red are parity double-buffered, so the
        // next write to this parity's buffers (step t+2) is ordered behind
        // barrier2(t+1), which postdates every step-t read.
    }

    // ---- finish: alpha += trans[n, END]; denominator = logsumexp_n(alpha) ----
#pragma unroll
    for (int vr = 0; vr < 8; ++vr) D[vr] += trans[n * TROW + (NTg + 1)];

    float pm[8];
#pragma unroll
    for (int vr = 0; vr < 8; ++vr) {
        float v = D[vr];
        v = fmaxf(v, __shfl_xor(v, 1));
        v = fmaxf(v, __shfl_xor(v, 2));
        v = fmaxf(v, __shfl_xor(v, 4));
        v = fmaxf(v, __shfl_xor(v, 8));
        pm[vr] = v;
    }
    if (lcol == 0) {
#pragma unroll
        for (int vr = 0; vr < 8; ++vr) Red[0][vr + 8 * half][w] = pm[vr];
    }
    __syncthreads();
    float mx[8];
#pragma unroll
    for (int vr = 0; vr < 8; ++vr) {
        v4f q = *(const v4f*)&Red[0][vr + 8 * half][0];
        mx[vr] = fmaxf(fmaxf(q[0], q[1]), fmaxf(q[2], q[3]));
    }
    __syncthreads();
#pragma unroll
    for (int vr = 0; vr < 8; ++vr) {
        float v = __builtin_amdgcn_exp2f((D[vr] - mx[vr]) * CRF_LOG2E);
        v += __shfl_xor(v, 1);
        v += __shfl_xor(v, 2);
        v += __shfl_xor(v, 4);
        v += __shfl_xor(v, 8);
        pm[vr] = v;
    }
    if (lcol == 0) {
#pragma unroll
        for (int vr = 0; vr < 8; ++vr) Red[0][vr + 8 * half][w] = pm[vr];
    }
    __syncthreads();
    if (w == 0 && lcol == 0) {
#pragma unroll
        for (int vr = 0; vr < 8; ++vr) {
            v4f q = *(const v4f*)&Red[0][vr + 8 * half][0];
            float s = (q[0] + q[1]) + (q[2] + q[3]);   // s >= 1, raw v_log safe
            den[bbase + vr + 8 * half] = mx[vr] + CRF_LN2 * __builtin_amdgcn_logf(s);
        }
    }
}

// ---------------------------------------------------------------------------
// Kernel 3: out = mean(denominator - numerator)
// ---------------------------------------------------------------------------
__global__ __launch_bounds__(512)
void CRF_final_kernel(const float* __restrict__ den,
                      const float* __restrict__ num,
                      float* __restrict__ out) {
    __shared__ float red[512];
    const int tid = threadIdx.x;
    red[tid] = den[tid] - num[tid];
    __syncthreads();
    for (int s2 = 256; s2 > 0; s2 >>= 1) {
        if (tid < s2) red[tid] += red[tid + s2];
        __syncthreads();
    }
    if (tid == 0) out[0] = red[0] * (1.0f / (float)Bdim);
}

// ---------------------------------------------------------------------------
extern "C" void kernel_launch(void* const* d_in, const int* in_sizes, int n_in,
                              void* d_out, int out_size, void* d_ws, size_t ws_size,
                              hipStream_t stream) {
    const float*         em    = (const float*)d_in[0];          // (512,1024,64) f32
    const int*           tags  = (const int*)d_in[1];            // (512,1024) int
    const unsigned char* mask  = (const unsigned char*)d_in[2];  // (512,1024) bool
    const float*         trans = (const float*)d_in[3];          // (66,66) f32
    float* out = (float*)d_out;

    float* ws_num = (float*)d_ws;          // [512]
    float* ws_den = ws_num + Bdim;         // [512]

    CRF_numer_kernel<<<Bdim, 256, 0, stream>>>(em, tags, mask, trans, ws_num);
    CRF_forward_kernel<<<Bdim / 16, 128, 0, stream>>>(em, trans, mask, ws_den);
    CRF_final_kernel<<<1, 512, 0, stream>>>(ws_den, ws_num, out);
}
